// ChannelDiffusion_68968584839219
// MI455X (gfx1250) — matrix operand order repaired and checked by reading, not compiled, not run on hardware
//
#include <hip/hip_runtime.h>
#include <hip/hip_bf16.h>

// ChannelDiffusion on MI455X (gfx1250): bf16 WMMA pipeline, fp32 accumulate.
// B=2, N=8192, D=1024, H=16, dh=64.
// Double-buffered LDS pipeline; async global->LDS staging on the bf16-A GEMM.

typedef __bf16 bf16_t;
typedef __attribute__((ext_vector_type(2)))  __bf16 v2bf;
typedef __attribute__((ext_vector_type(8)))  __bf16 v8bf;
typedef __attribute__((ext_vector_type(16))) __bf16 v16bf;
typedef __attribute__((ext_vector_type(8)))  float  v8f;
typedef __attribute__((ext_vector_type(4)))  float  v4f;

#define BB 2
#define NN 8192
#define DD 1024
#define HH 16
#define DH 64
#define BK 64

__device__ __forceinline__ v8f wmma_bf(v16bf a, v16bf b, v8f c) {
  // D = A(16x32 bf16) * B(32x16 bf16) + C(16x16 f32)
  return __builtin_amdgcn_wmma_f32_16x16x32_bf16(
      false, a, false, b, (short)0, c, false, false);
}

__device__ __forceinline__ v16bf frag_cat(v8bf lo, v8bf hi) {
  return __builtin_shufflevector(lo, hi, 0,1,2,3,4,5,6,7,8,9,10,11,12,13,14,15);
}

__device__ __forceinline__ v8f vzero8() {
  v8f z = {0.f,0.f,0.f,0.f,0.f,0.f,0.f,0.f};
  return z;
}

// CDNA5 async global->LDS (ASYNCcnt-tracked, GV mode), 16B per lane.
__device__ __forceinline__ void async_load_b128(unsigned lds_off, const void* gaddr) {
  unsigned long long ga = (unsigned long long)(size_t)gaddr;
  asm volatile("global_load_async_to_lds_b128 %0, %1, off"
               :: "v"(lds_off), "v"(ga) : "memory");
}
__device__ __forceinline__ void wait_async0() {
  asm volatile("s_wait_asynccnt 0x0" ::: "memory");
}

// A-fragment (16x32 bf16): lane l (m=l&15, half=l>>4): row m,
// K = ks + {half*8 .. +8} and ks+16 + {half*8 .. +8}   (ISA 7.12.2 layout)
template<int LDT>
__device__ __forceinline__ v16bf lds_frag_a(const bf16_t* base, int rowbase,
                                            int ks, int lane) {
  const int m = lane & 15, half = lane >> 4;
  const v8bf* p0 = (const v8bf*)(base + (rowbase + m) * LDT + ks + half * 8);
  const v8bf* p1 = (const v8bf*)(base + (rowbase + m) * LDT + ks + 16 + half * 8);
  return frag_cat(*p0, *p1);
}

// B-fragment (32x16 bf16) from K-transposed tile Bt[n][k]: lane l (n=l&15,
// half=l>>4): column n, K = ks + {half*8 .. +8} and ks+16 + {half*8 .. +8}.
template<int LDT>
__device__ __forceinline__ v16bf lds_frag_b(const bf16_t* base, int colbase,
                                            int ks, int lane) {
  const int n = lane & 15, half = lane >> 4;
  const v8bf* p0 = (const v8bf*)(base + (colbase + n) * LDT + ks + half * 8);
  const v8bf* p1 = (const v8bf*)(base + (colbase + n) * LDT + ks + 16 + half * 8);
  return frag_cat(*p0, *p1);
}

// ---------------------------------------------------------------------------
__global__ __launch_bounds__(256) void cvt_f32_bf16(const float* __restrict__ src,
                                                    bf16_t* __restrict__ dst, int n) {
  int i = blockIdx.x * blockDim.x + threadIdx.x;
  if (i < n) dst[i] = (bf16_t)src[i];
}

// ---------------------------------------------------------------------------
// Pipelined WMMA GEMM: C[M,N] = A[M,K]*B[K,N]. BM=BN=128, BK=64, 4 waves/WG,
// wave tile 64x64 (4x4 accumulators), double-buffered LDS.
//  - A bf16: staged via async global->LDS directly into the inactive buffer.
//  - A f32 : loaded, converted, stored (VALU cvt) into the inactive buffer.
//  - B     : loaded to regs (overlaps compute), stored K-pair-packed (b32)
//            transposed into Bt[n][k].
// ---------------------------------------------------------------------------
template<bool OUT_F32, bool A_F32>
__global__ __launch_bounds__(128) void gemm_wmma(
    const void* __restrict__ Aptr, const bf16_t* __restrict__ B,
    void* __restrict__ Cptr, int lda, int ldb, int ldc, int K)
{
  constexpr int LDT = BK + 8;  // 72 elems (144B rows: 16B-aligned chunks)
  __shared__ __align__(16) bf16_t sA[2][128 * LDT];
  __shared__ __align__(16) bf16_t sB[2][128 * LDT];

  const int tid = threadIdx.x, lane = tid & 31, wave = tid >> 5;
  const int wr = (wave >> 1) * 64, wc = (wave & 1) * 64;
  const int m0 = blockIdx.y * 128, n0 = blockIdx.x * 128;

  // ---- A staging: one 64-elem row per thread ----
  auto stageA = [&](int k0, int buf) {
    if constexpr (A_F32) {
      const float* a = (const float*)Aptr + (size_t)(m0 + tid) * lda + k0;
      bf16_t* dst = &sA[buf][tid * LDT];
      #pragma unroll
      for (int e = 0; e < BK; e += 4) {
        v4f v = *(const v4f*)(a + e);
        v2bf p0 = {(bf16_t)v.x, (bf16_t)v.y};
        v2bf p1 = {(bf16_t)v.z, (bf16_t)v.w};
        *(v2bf*)(dst + e)     = p0;
        *(v2bf*)(dst + e + 2) = p1;
      }
    } else {
      const bf16_t* a = (const bf16_t*)Aptr + (size_t)(m0 + tid) * lda + k0;
      const unsigned l0 = (unsigned)(size_t)&sA[buf][tid * LDT];
      #pragma unroll
      for (int c = 0; c < BK / 8; ++c)
        async_load_b128(l0 + c * 16, a + c * 8);
    }
  };

  // ---- B staging: thread owns k-pair (kp,kp+1) x 32 columns ----
  const int kp = (tid & 31) * 2;
  const int nc = (tid >> 5) * 32;
  v8bf br0[4], br1[4];
  auto loadB = [&](int k0) {
    const bf16_t* b0 = B + (size_t)(k0 + kp) * ldb + n0 + nc;
    const bf16_t* b1 = b0 + ldb;
    #pragma unroll
    for (int e = 0; e < 4; ++e) {
      br0[e] = *(const v8bf*)(b0 + e * 8);
      br1[e] = *(const v8bf*)(b1 + e * 8);
    }
    __builtin_prefetch(b0 + (size_t)2 * BK * ldb, 0, 0);  // global_prefetch_b8
  };
  auto storeB = [&](int buf) {
    #pragma unroll
    for (int e = 0; e < 32; ++e) {
      v2bf pk = {br0[e >> 3][e & 7], br1[e >> 3][e & 7]};
      *(v2bf*)&sB[buf][(nc + e) * LDT + kp] = pk;  // b32 packed transpose
    }
  };

  v8f acc[4][4];
  #pragma unroll
  for (int i = 0; i < 4; ++i)
    #pragma unroll
    for (int j = 0; j < 4; ++j) acc[i][j] = vzero8();

  auto compute = [&](int buf) {
    #pragma unroll
    for (int ks = 0; ks < BK; ks += 32) {
      v16bf af[4], bfr[4];
      #pragma unroll
      for (int i = 0; i < 4; ++i) af[i]  = lds_frag_a<LDT>(sA[buf], wr + 16*i, ks, lane);
      #pragma unroll
      for (int j = 0; j < 4; ++j) bfr[j] = lds_frag_b<LDT>(sB[buf], wc + 16*j, ks, lane);
      #pragma unroll
      for (int i = 0; i < 4; ++i)
        #pragma unroll
        for (int j = 0; j < 4; ++j)
          acc[i][j] = wmma_bf(af[i], bfr[j], acc[i][j]);
    }
  };

  // ---- prologue ----
  loadB(0);
  stageA(0, 0);
  storeB(0);
  if constexpr (!A_F32) wait_async0();
  __syncthreads();

  // ---- pipelined main loop: one barrier per BK ----
  int p = 0;
  for (int k0 = 0; k0 < K; k0 += BK) {
    const int nxt = k0 + BK;
    if (nxt < K) { loadB(nxt); stageA(nxt, p ^ 1); }
    compute(p);
    if (nxt < K) storeB(p ^ 1);
    if constexpr (!A_F32) wait_async0();
    __syncthreads();
    p ^= 1;
  }

  // ---- epilogue: D layout — VGPR r: (M = base+r | base+8+r, N = lane&15) ----
  const int nlo = lane & 15, mhi = (lane >> 4) * 8;
  #pragma unroll
  for (int i = 0; i < 4; ++i) {
    #pragma unroll
    for (int j = 0; j < 4; ++j) {
      const int row = m0 + wr + 16 * i + mhi;
      const int col = n0 + wc + 16 * j + nlo;
      #pragma unroll
      for (int r = 0; r < 8; ++r) {
        const size_t idx = (size_t)(row + r) * ldc + col;
        if constexpr (OUT_F32) ((float*)Cptr)[idx] = acc[i][j][r];
        else                   ((bf16_t*)Cptr)[idx] = (bf16_t)acc[i][j][r];
      }
    }
  }
}

// ---------------------------------------------------------------------------
// Gram + softmax per (b,h): G = Q^T Q (K = 8192), logits =
// (2G - d_c - d_d)/sqrt(N) * tau[h], row softmax -> attn bf16 (row-major).
// 8 waves; wave owns 2 of 16 output tiles; Q staged K-transposed (pair-packed
// b32 stores), double-buffered, BK=64.
// ---------------------------------------------------------------------------
__global__ __launch_bounds__(256) void gram_softmax(
    const bf16_t* __restrict__ qk, const float* __restrict__ tau,
    bf16_t* __restrict__ attn)
{
  constexpr int LDT = BK + 8;
  __shared__ __align__(16) bf16_t Qt[2][64 * LDT];  // Qt[c][k]
  __shared__ float G[64 * 68];
  __shared__ float diag[64];

  const int bh = blockIdx.x;
  const int b = bh >> 4, h = bh & 15;
  const int tid = threadIdx.x, lane = tid & 31, wave = tid >> 5;
  const bf16_t* Q = qk + (size_t)b * NN * DD + h * DH;

  const int i0 = ((wave * 2) >> 2) * 16;   // tiles 2w, 2w+1 share row block
  const int j0 = ((wave * 2) & 3) * 16;
  const int j1 = j0 + 16;

  const int kp = (tid & 31) * 2;           // k-pair
  const int cc = (tid >> 5) * 8;           // 8 channels
  v8bf q0, q1;
  auto loadQ = [&](int kk) {
    const bf16_t* p0 = Q + (size_t)(kk + kp) * DD + cc;
    q0 = *(const v8bf*)p0;
    q1 = *(const v8bf*)(p0 + DD);
  };
  auto storeQ = [&](int buf) {
    #pragma unroll
    for (int e = 0; e < 8; ++e) {
      v2bf pk = {q0[e], q1[e]};
      *(v2bf*)&Qt[buf][(cc + e) * LDT + kp] = pk;
    }
  };

  v8f acc0 = vzero8(), acc1 = vzero8();
  auto compute = [&](int buf) {
    #pragma unroll
    for (int ks = 0; ks < BK; ks += 32) {
      v16bf a0 = lds_frag_a<LDT>(Qt[buf], i0, ks, lane);  // A[c,k] = Q[k,c]
      v16bf b0 = lds_frag_b<LDT>(Qt[buf], j0, ks, lane);  // B[k,d] = Q[k,d]
      v16bf b1 = lds_frag_b<LDT>(Qt[buf], j1, ks, lane);
      acc0 = wmma_bf(a0, b0, acc0);
      acc1 = wmma_bf(a0, b1, acc1);
    }
  };

  loadQ(0); storeQ(0); __syncthreads();
  int p = 0;
  for (int kk = 0; kk < NN; kk += BK) {
    const int nxt = kk + BK;
    if (nxt < NN) loadQ(nxt);
    compute(p);
    if (nxt < NN) storeQ(p ^ 1);
    __syncthreads();
    p ^= 1;
  }

  const int nlo = lane & 15, mhi = (lane >> 4) * 8;
  #pragma unroll
  for (int r = 0; r < 8; ++r) {
    G[(i0 + mhi + r) * 68 + j0 + nlo] = acc0[r];
    G[(i0 + mhi + r) * 68 + j1 + nlo] = acc1[r];
  }
  __syncthreads();
  if (tid < 64) diag[tid] = G[tid * 68 + tid];
  __syncthreads();

  if (tid < 64) {
    const float inv = 0.011048543456039806f;  // 1/sqrt(8192)
    const float th  = tau[h];
    const int c = tid;
    const float qc = diag[c];
    float mx = -3.4e38f;
    for (int d = 0; d < 64; ++d) {
      float l = (2.f * G[c * 68 + d] - qc - diag[d]) * inv * th;
      G[c * 68 + d] = l;
      mx = fmaxf(mx, l);
    }
    float s = 0.f;
    for (int d = 0; d < 64; ++d) {
      float e = __expf(G[c * 68 + d] - mx);
      G[c * 68 + d] = e;
      s += e;
    }
    const float rs = 1.f / s;
    bf16_t* arow = attn + (size_t)bh * (DH * DH) + c * DH;
    for (int d = 0; d < 64; ++d) arow[d] = (bf16_t)(G[c * 68 + d] * rs);
  }
}

// ---------------------------------------------------------------------------
// u[b,n,h*64+c] = sum_d v[b,n,h*64+d] * attn[b,h][c,d].  B[k=d][n=c] column c
// == attn row c (contiguous), A rows contiguous: all fragments straight from
// global, K=64 fully unrolled. 4 waves/WG, wave tile 64x64.
// ---------------------------------------------------------------------------
__global__ __launch_bounds__(128) void apply_attn(
    const bf16_t* __restrict__ v16, const bf16_t* __restrict__ attn,
    bf16_t* __restrict__ u16)
{
  const int bh = blockIdx.y;
  const int b = bh >> 4, h = bh & 15;
  const int tid = threadIdx.x, lane = tid & 31, wave = tid >> 5;
  const int m0 = blockIdx.x * 256 + wave * 64;

  const bf16_t* Abase = v16 + ((size_t)b * NN + m0) * DD + h * DH;
  const bf16_t* Bbase = attn + (size_t)bh * (DH * DH);

  const int nlo = lane & 15, half = lane >> 4;

  v8f acc[4][4];
  #pragma unroll
  for (int i = 0; i < 4; ++i)
    #pragma unroll
    for (int j = 0; j < 4; ++j) acc[i][j] = vzero8();

  #pragma unroll
  for (int k0 = 0; k0 < DH; k0 += 32) {
    v16bf af[4], bfr[4];
    #pragma unroll
    for (int i = 0; i < 4; ++i) {
      const bf16_t* p = Abase + (size_t)(16 * i + nlo) * DD + k0 + half * 8;
      af[i] = frag_cat(*(const v8bf*)p, *(const v8bf*)(p + 16));
    }
    #pragma unroll
    for (int j = 0; j < 4; ++j) {
      const bf16_t* p = Bbase + (16 * j + nlo) * DH + k0 + half * 8;
      bfr[j] = frag_cat(*(const v8bf*)p, *(const v8bf*)(p + 16));
    }
    #pragma unroll
    for (int i = 0; i < 4; ++i)
      #pragma unroll
      for (int j = 0; j < 4; ++j)
        acc[i][j] = wmma_bf(af[i], bfr[j], acc[i][j]);
  }

  const int mhi = half * 8;
  #pragma unroll
  for (int i = 0; i < 4; ++i) {
    #pragma unroll
    for (int j = 0; j < 4; ++j) {
      #pragma unroll
      for (int r = 0; r < 8; ++r) {
        const size_t idx =
            ((size_t)b * NN + m0 + 16 * i + mhi + r) * DD + h * DH + 16 * j + nlo;
        u16[idx] = (bf16_t)acc[i][j][r];
      }
    }
  }
}

// ---------------------------------------------------------------------------
extern "C" void kernel_launch(void* const* d_in, const int* in_sizes, int n_in,
                              void* d_out, int out_size, void* d_ws, size_t ws_size,
                              hipStream_t stream)
{
  const float* x    = (const float*)d_in[0];
  const float* Wqk  = (const float*)d_in[1];
  const float* Wv   = (const float*)d_in[2];
  const float* Wout = (const float*)d_in[3];
  const float* tau  = (const float*)d_in[4];

  char* ws = (char*)d_ws;
  const size_t SZ_MAT = (size_t)BB * NN * DD * sizeof(bf16_t);  // 32 MB
  const size_t SZ_W   = (size_t)DD * DD * sizeof(bf16_t);       //  2 MB
  bf16_t* qk16   = (bf16_t*)(ws);
  bf16_t* v16b   = (bf16_t*)(ws + SZ_MAT);
  bf16_t* wqk16  = (bf16_t*)(ws + 2 * SZ_MAT);
  bf16_t* wv16   = (bf16_t*)(ws + 2 * SZ_MAT + SZ_W);
  bf16_t* wout16 = (bf16_t*)(ws + 2 * SZ_MAT + 2 * SZ_W);
  bf16_t* attn16 = (bf16_t*)(ws + 2 * SZ_MAT + 3 * SZ_W);
  bf16_t* u16    = qk16;  // qk16 dead after gram_softmax; reuse as u

  const int nw = DD * DD;
  cvt_f32_bf16<<<(nw + 255) / 256, 256, 0, stream>>>(Wqk,  wqk16,  nw);
  cvt_f32_bf16<<<(nw + 255) / 256, 256, 0, stream>>>(Wv,   wv16,   nw);
  cvt_f32_bf16<<<(nw + 255) / 256, 256, 0, stream>>>(Wout, wout16, nw);

  const dim3 gGemm(DD / 128, (BB * NN) / 128);
  gemm_wmma<false, true><<<gGemm, 128, 0, stream>>>(
      x, wqk16, qk16, DD, DD, DD, DD);
  gemm_wmma<false, true><<<gGemm, 128, 0, stream>>>(
      x, wv16, v16b, DD, DD, DD, DD);

  gram_softmax<<<BB * HH, 256, 0, stream>>>(qk16, tau, attn16);

  apply_attn<<<dim3(NN / 256, BB * HH), 128, 0, stream>>>(v16b, attn16, u16);

  gemm_wmma<true, false><<<gGemm, 128, 0, stream>>>(
      u16, wout16, (float*)d_out, DD, DD, DD, DD);
}
